// NN_each_LN_exp_55765855371904
// MI455X (gfx1250) — compile-verified, matrix operand order checked by reading, not run on hardware
//
#include <hip/hip_runtime.h>

// ---------------------------------------------------------------------------
// Problem constants
// ---------------------------------------------------------------------------
#define BATCH   4096
#define PIX     1024          // (V-2)*H = 32*32
#define KS      33
#define NKIND   4
#define NDEPTH  4
#define NSQ     100
#define NELEM   (BATCH * PIX)   // 4 M elements per map
#define NPAIR   (NELEM / 2)     // uint32 (bf16-pair) count per swizzled map

typedef __attribute__((ext_vector_type(16))) __bf16 v16bf;
typedef __attribute__((ext_vector_type(8)))  float  v8f;
typedef __attribute__((ext_vector_type(8)))  int    v8i;

union U8I16BF { v8i i; v16bf b; };
__device__ __forceinline__ v16bf as_bf(v8i x) { U8I16BF u; u.i = x; return u.b; }

// ---------------------------------------------------------------------------
// bf16 helpers (raw ushort storage; round-to-nearest-even)
// ---------------------------------------------------------------------------
__device__ __forceinline__ unsigned short f2bf(float f) {
  union { float f; unsigned int u; } v; v.f = f;
  unsigned int u = v.u;
  return (unsigned short)((u + 0x7FFFu + ((u >> 16) & 1u)) >> 16);
}
__device__ __forceinline__ unsigned int pack_bf(float lo, float hi) {
  return ((unsigned int)f2bf(hi) << 16) | (unsigned int)f2bf(lo);
}

// ---------------------------------------------------------------------------
// Fragment swizzle: 16x16x32 bf16 operand layout (ISA 7.12.2).
// Tile (rt, kt): flat[(((rt*32 + kt)*32) + lane)*8 + v], each uint32 packs
// bf16 pair {K, K+1} with K = kt*32 + kpack(v,hi), row/col = rt*16 + lane%16.
//   kpack(v,hi) = (v<4 ? 2v : 16+2(v-4)) + 8*hi
// Inverse (for producers): given even k-in-tile kp:
//   hi = (kp>>3)&1 ; rem = kp-8*hi ; v = rem<8 ? rem/2 : 4+(rem-16)/2
// ---------------------------------------------------------------------------
__device__ __forceinline__ size_t asw_index(int row, int p_even) {
  int kt  = p_even >> 5;
  int kp  = p_even & 31;
  int hi  = (kp >> 3) & 1;
  int rem = kp - hi * 8;
  int v   = (rem < 8) ? (rem >> 1) : (4 + ((rem - 16) >> 1));
  int rt  = row >> 4;
  int lane = hi * 16 + (row & 15);
  return (((size_t)(rt * 32 + kt) * 32) + lane) * 8 + v;
}

// ---------------------------------------------------------------------------
// 1) Conv kernel (33x33) -> dense 1024x1024 operand matrix, pre-swizzled to
//    the WMMA B-fragment layout.  W[k][c] = wk[(iy-oy+16)*33 + (ix-ox+16)].
// ---------------------------------------------------------------------------
__device__ __forceinline__ float conv_w(const float* wk, int k, int c) {
  int iy = k >> 5, ix = k & 31, oy = c >> 5, ox = c & 31;
  int dy = iy - oy + 16, dx = ix - ox + 16;
  if ((unsigned)dy > 32u || (unsigned)dx > 32u) return 0.0f;
  return wk[dy * KS + dx];
}

__global__ __launch_bounds__(256) void build_wsw(const float* __restrict__ w0,
                                                 const float* __restrict__ w1,
                                                 const float* __restrict__ w2,
                                                 const float* __restrict__ w3,
                                                 unsigned int* __restrict__ wsw) {
  int bid = blockIdx.x;                 // 4 mats * 64 nt * 32 kt = 8192 blocks
  int kt  = bid & 31;
  int nt  = (bid >> 5) & 63;
  int mat = bid >> 11;
  const float* wk = (mat == 0) ? w0 : (mat == 1) ? w1 : (mat == 2) ? w2 : w3;
  unsigned int* outp = wsw + (size_t)mat * NPAIR / 8;  // 524288 u32 per matrix

  int lane = threadIdx.x & 31;
  int v    = threadIdx.x >> 5;
  int n    = nt * 16 + (lane & 15);
  int hi   = lane >> 4;
  int kp   = ((v < 4) ? (2 * v) : (16 + 2 * (v - 4))) + hi * 8;
  int k0   = kt * 32 + kp;

  outp[((((nt * 32 + kt) * 32) + lane) << 3) + v] =
      pack_bf(conv_w(wk, k0, n), conv_w(wk, k0 + 1, n));
}

// ---------------------------------------------------------------------------
// 2) Core GEMM:  OUT_f32[BATCH x 1024] = A(swizzled bf16) @ W(swizzled bf16)
//    Block = 256 thr = 8 waves; block tile 128 rows x 128 cols.
//    Wave tile 32 rows x 64 cols = 8 accumulators = 8 WMMA per K-step,
//    6 contiguous 32B fragment loads per K-step, double-buffered so the
//    s_wait before each WMMA group tolerates in-flight next-tile loads.
// ---------------------------------------------------------------------------
__global__ __launch_bounds__(256) void gemm_conv(const unsigned int* __restrict__ A,
                                                 const unsigned int* __restrict__ Wsw,
                                                 float* __restrict__ OUT) {
  const int lane = threadIdx.x & 31;
  const int wave = threadIdx.x >> 5;
  const int wrow = wave & 3;                   // 4 row groups x 32 rows
  const int wcol = wave >> 2;                  // 2 col groups x 64 cols
  const int mt0  = blockIdx.x * 8 + wrow * 2;  // first of 2 row tiles
  const int colbase = blockIdx.y * 128 + wcol * 64;
  const int ntbase  = colbase >> 4;
  const int m  = lane & 15;
  const int hi = lane >> 4;

  const unsigned int* Ap0 = A + (((size_t)mt0 * 32) * 32 + lane) * 8;
  const unsigned int* Ap1 = Ap0 + 32 * 32 * 8;        // next 16-row tile
  const unsigned int* Bp0 = Wsw + (((size_t)(ntbase + 0) * 32) * 32 + lane) * 8;
  const unsigned int* Bp1 = Wsw + (((size_t)(ntbase + 1) * 32) * 32 + lane) * 8;
  const unsigned int* Bp2 = Wsw + (((size_t)(ntbase + 2) * 32) * 32 + lane) * 8;
  const unsigned int* Bp3 = Wsw + (((size_t)(ntbase + 3) * 32) * 32 + lane) * 8;

  v8f acc[2][4];
#pragma unroll
  for (int t = 0; t < 2; ++t)
#pragma unroll
    for (int j = 0; j < 4; ++j)
#pragma unroll
      for (int r = 0; r < 8; ++r) acc[t][j][r] = 0.0f;

  // Prologue: load K-tile 0 fragments (one 32B contiguous load each)
  v8i a0 = *(const v8i*)Ap0;
  v8i a1 = *(const v8i*)Ap1;
  v8i b0 = *(const v8i*)Bp0;
  v8i b1 = *(const v8i*)Bp1;
  v8i b2 = *(const v8i*)Bp2;
  v8i b3 = *(const v8i*)Bp3;

  for (int kt = 0; kt < 32; ++kt) {
    v8i na0 = a0, na1 = a1, nb0 = b0, nb1 = b1, nb2 = b2, nb3 = b3;
    if (kt < 31) {                       // issue next-tile loads first
      const size_t s = (size_t)(kt + 1) * 256;   // 32 lanes * 8 u32 per tile
      na0 = *(const v8i*)(Ap0 + s);
      na1 = *(const v8i*)(Ap1 + s);
      nb0 = *(const v8i*)(Bp0 + s);
      nb1 = *(const v8i*)(Bp1 + s);
      nb2 = *(const v8i*)(Bp2 + s);
      nb3 = *(const v8i*)(Bp3 + s);
    }
    v16bf av0 = as_bf(a0), av1 = as_bf(a1);
    acc[0][0] = __builtin_amdgcn_wmma_f32_16x16x32_bf16(false, av0, false, as_bf(b0), (short)0, acc[0][0], false, false);
    acc[1][0] = __builtin_amdgcn_wmma_f32_16x16x32_bf16(false, av1, false, as_bf(b0), (short)0, acc[1][0], false, false);
    acc[0][1] = __builtin_amdgcn_wmma_f32_16x16x32_bf16(false, av0, false, as_bf(b1), (short)0, acc[0][1], false, false);
    acc[1][1] = __builtin_amdgcn_wmma_f32_16x16x32_bf16(false, av1, false, as_bf(b1), (short)0, acc[1][1], false, false);
    acc[0][2] = __builtin_amdgcn_wmma_f32_16x16x32_bf16(false, av0, false, as_bf(b2), (short)0, acc[0][2], false, false);
    acc[1][2] = __builtin_amdgcn_wmma_f32_16x16x32_bf16(false, av1, false, as_bf(b2), (short)0, acc[1][2], false, false);
    acc[0][3] = __builtin_amdgcn_wmma_f32_16x16x32_bf16(false, av0, false, as_bf(b3), (short)0, acc[0][3], false, false);
    acc[1][3] = __builtin_amdgcn_wmma_f32_16x16x32_bf16(false, av1, false, as_bf(b3), (short)0, acc[1][3], false, false);
    a0 = na0; a1 = na1; b0 = nb0; b1 = nb1; b2 = nb2; b3 = nb3;
  }

  // C/D layout: VGPR r -> M=r (lanes 0-15) / M=8+r (lanes 16-31), N=lane%16
#pragma unroll
  for (int t = 0; t < 2; ++t) {
    int rowbase = (mt0 + t) * 16 + hi * 8;
#pragma unroll
    for (int j = 0; j < 4; ++j) {
      int ocol = colbase + j * 16 + m;
#pragma unroll
      for (int r = 0; r < 8; ++r)
        OUT[(size_t)(rowbase + r) * PIX + ocol] = acc[t][j][r];
    }
  }
}

// ---------------------------------------------------------------------------
// 3) Masks from the int board, written directly in swizzled fragment layout.
//    dots layout: (V=34,H=32,B) -> flat (y*32+x)*4096 + b for y < 32.
//    One thread handles a bf16 pair (pixels 2q, 2q+1) of one board row.
// ---------------------------------------------------------------------------
__global__ __launch_bounds__(256) void build_masks(const int* __restrict__ dots,
                                                   unsigned int* __restrict__ empty_sw,
                                                   unsigned int* __restrict__ each_sw,
                                                   unsigned int* __restrict__ notm_sw) {
  int idx = blockIdx.x * 256 + threadIdx.x;        // over NPAIR
  if (idx >= NPAIR) return;
  int q = idx & 511;
  int b = idx >> 9;
  int p0 = q * 2;
  int d0 = dots[(size_t)p0 * BATCH + b];
  int d1 = dots[(size_t)(p0 + 1) * BATCH + b];
  size_t o = asw_index(b, p0);
  const unsigned int LO = 0x3F80u, HI = 0x3F800000u;
  empty_sw[o] = (d0 == 0 ? LO : 0u) | (d1 == 0 ? HI : 0u);
#pragma unroll
  for (int k = 0; k < NKIND; ++k) {
    each_sw[(size_t)k * NPAIR + o] = (d0 == k + 1 ? LO : 0u) | (d1 == k + 1 ? HI : 0u);
    notm_sw[(size_t)k * NPAIR + o] =
        ((d0 != 0 && d0 != k + 1) ? LO : 0u) | ((d1 != 0 && d1 != k + 1) ? HI : 0u);
  }
}

// ---------------------------------------------------------------------------
// 4) First fused pass: L[k]=each[k]*(conv_each[k]+E-notm_conv[k]);
//    s=sigmoid(sum L); t[k]=bf16(notm_conv[k]*s) written swizzled.
//    `each` recomputed from dots (1 int load replaces 4 mask loads).
// ---------------------------------------------------------------------------
__global__ __launch_bounds__(256) void first_combine(const int* __restrict__ dots,
                                                     const float* __restrict__ E,
                                                     const float* __restrict__ CE,
                                                     const float* __restrict__ NC,
                                                     float* __restrict__ L,
                                                     float* __restrict__ S,
                                                     unsigned int* __restrict__ T_sw) {
  int idx = blockIdx.x * 256 + threadIdx.x;        // over NPAIR
  if (idx >= NPAIR) return;
  int q = idx & 511;
  int b = idx >> 9;
  int p0 = q * 2;
  size_t lin = (size_t)b * PIX + p0;
  size_t o   = asw_index(b, p0);
  int d0 = dots[(size_t)p0 * BATCH + b];
  int d1 = dots[(size_t)(p0 + 1) * BATCH + b];
  float e0 = E[lin], e1 = E[lin + 1];
  float nc0[NKIND], nc1[NKIND];
  float s0 = 0.0f, s1 = 0.0f;
#pragma unroll
  for (int k = 0; k < NKIND; ++k) {
    size_t off = (size_t)k * NELEM + lin;
    nc0[k] = NC[off]; nc1[k] = NC[off + 1];
    float L0 = (d0 == k + 1) ? (CE[off] + e0 - nc0[k]) : 0.0f;
    float L1 = (d1 == k + 1) ? (CE[off + 1] + e1 - nc1[k]) : 0.0f;
    L[off] = L0; L[off + 1] = L1;
    s0 += L0; s1 += L1;
  }
  s0 = 1.0f / (1.0f + __expf(-s0));
  s1 = 1.0f / (1.0f + __expf(-s1));
  S[lin] = s0; S[lin + 1] = s1;
#pragma unroll
  for (int k = 0; k < NKIND; ++k)
    T_sw[(size_t)k * NPAIR + o] = pack_bf(nc0[k] * s0, nc1[k] * s1);
}

// ---------------------------------------------------------------------------
// 5) Depth pass: L[k]=each[k]*(L[k]+E+C[k]); s=sigmoid(sum); t[k]=bf16(nc*s)
// ---------------------------------------------------------------------------
__global__ __launch_bounds__(256) void depth_combine(const int* __restrict__ dots,
                                                     const float* __restrict__ E,
                                                     const float* __restrict__ C,
                                                     const float* __restrict__ NC,
                                                     float* __restrict__ L,
                                                     float* __restrict__ S,
                                                     unsigned int* __restrict__ T_sw) {
  int idx = blockIdx.x * 256 + threadIdx.x;        // over NPAIR
  if (idx >= NPAIR) return;
  int q = idx & 511;
  int b = idx >> 9;
  int p0 = q * 2;
  size_t lin = (size_t)b * PIX + p0;
  size_t o   = asw_index(b, p0);
  int d0 = dots[(size_t)p0 * BATCH + b];
  int d1 = dots[(size_t)(p0 + 1) * BATCH + b];
  float e0 = E[lin], e1 = E[lin + 1];
  float s0 = 0.0f, s1 = 0.0f;
#pragma unroll
  for (int k = 0; k < NKIND; ++k) {
    size_t off = (size_t)k * NELEM + lin;
    float L0 = (d0 == k + 1) ? (L[off] + e0 + C[off]) : 0.0f;
    float L1 = (d1 == k + 1) ? (L[off + 1] + e1 + C[off + 1]) : 0.0f;
    L[off] = L0; L[off + 1] = L1;
    s0 += L0; s1 += L1;
  }
  s0 = 1.0f / (1.0f + __expf(-s0));
  s1 = 1.0f / (1.0f + __expf(-s1));
  S[lin] = s0; S[lin + 1] = s1;
#pragma unroll
  for (int k = 0; k < NKIND; ++k) {
    size_t off = (size_t)k * NELEM + lin;
    T_sw[(size_t)k * NPAIR + o] = pack_bf(NC[off] * s0, NC[off + 1] * s1);
  }
}

// ---------------------------------------------------------------------------
// 6) Tiny MLP tail (~0.4 G MACs, negligible)
// ---------------------------------------------------------------------------
__global__ __launch_bounds__(256) void mlp1(const float* __restrict__ S,
                                            const float* __restrict__ W1,
                                            float* __restrict__ X1) {
  int idx = blockIdx.x * 256 + threadIdx.x;
  if (idx >= BATCH * NSQ) return;
  int b = idx / NSQ, j = idx % NSQ;
  const float* s = S + (size_t)b * PIX;
  const float* w = W1 + (size_t)j * PIX;
  float acc = 0.0f;
  for (int p = 0; p < PIX; ++p) acc = fmaf(s[p], w[p], acc);
  X1[idx] = (acc > 0.0f) ? acc : 0.2f * acc;
}

__global__ __launch_bounds__(256) void mlp2(const float* __restrict__ X1,
                                            const float* __restrict__ W2,
                                            float* __restrict__ X2) {
  int idx = blockIdx.x * 256 + threadIdx.x;
  if (idx >= BATCH * NSQ) return;
  int b = idx / NSQ, j = idx % NSQ;
  const float* x = X1 + (size_t)b * NSQ;
  const float* w = W2 + (size_t)j * NSQ;
  float acc = 0.0f;
  for (int i = 0; i < NSQ; ++i) acc = fmaf(x[i], w[i], acc);
  X2[idx] = (acc > 0.0f) ? acc : 0.2f * acc;
}

__global__ __launch_bounds__(256) void mlp3(const float* __restrict__ X2,
                                            const float* __restrict__ W3,
                                            float* __restrict__ OUT) {
  int b = blockIdx.x * 256 + threadIdx.x;
  if (b >= BATCH) return;
  const float* x = X2 + (size_t)b * NSQ;
  float acc = 0.0f;
  for (int j = 0; j < NSQ; ++j) acc = fmaf(x[j], W3[j], acc);
  OUT[b] = acc;
}

// ---------------------------------------------------------------------------
// Host-side orchestration (all on `stream`, no allocation / sync)
// ---------------------------------------------------------------------------
extern "C" void kernel_launch(void* const* d_in, const int* in_sizes, int n_in,
                              void* d_out, int out_size, void* d_ws, size_t ws_size,
                              hipStream_t stream) {
  (void)in_sizes; (void)n_in; (void)out_size; (void)ws_size;

  const int*   dots    = (const int*)d_in[0];
  const float* w_each  = (const float*)d_in[1];
  const float* w_not   = (const float*)d_in[2];
  const float* w_not2  = (const float*)d_in[3];
  const float* w_empty = (const float*)d_in[4];
  const float* W1      = (const float*)d_in[5];
  const float* W2      = (const float*)d_in[6];
  const float* W3      = (const float*)d_in[7];
  float* out           = (float*)d_out;

  // Workspace layout (bytes)
  const size_t MB = 1024ull * 1024ull;
  char* ws = (char*)d_ws;
  unsigned int* WSW      = (unsigned int*)(ws + 0);          //   8 MB
  unsigned int* EMPTY_SW = (unsigned int*)(ws + 8 * MB);     //   8 MB
  unsigned int* EACH_SW  = (unsigned int*)(ws + 16 * MB);    //  32 MB
  unsigned int* NOTM_SW  = (unsigned int*)(ws + 48 * MB);    //  32 MB
  float*        E_F32    = (float*)(ws + 80 * MB);           //  16 MB
  float*        CE_F32   = (float*)(ws + 96 * MB);           //  64 MB (reused as C)
  float*        NC_F32   = (float*)(ws + 160 * MB);          //  64 MB
  float*        L_F32    = (float*)(ws + 224 * MB);          //  64 MB
  float*        S_F32    = (float*)(ws + 288 * MB);          //  16 MB
  unsigned int* T_SW     = (unsigned int*)(ws + 304 * MB);   //  32 MB
  float*        X1       = (float*)(ws + 336 * MB);          // 1.6 MB
  float*        X2       = (float*)(ws + 338 * MB);          // 1.6 MB

  const size_t MAT_U32 = (size_t)NPAIR / 8;  // 524288 u32 per swizzled conv mat

  build_wsw<<<dim3(4 * 64 * 32), dim3(256), 0, stream>>>(
      w_each, w_not, w_not2, w_empty, WSW);

  build_masks<<<dim3(NPAIR / 256), dim3(256), 0, stream>>>(
      dots, EMPTY_SW, EACH_SW, NOTM_SW);

  const dim3 ggrid(BATCH / 128, PIX / 128);
  const dim3 gblk(256);

  // E = empty @ W_empty (mat 3)
  gemm_conv<<<ggrid, gblk, 0, stream>>>(EMPTY_SW, WSW + 3 * MAT_U32, E_F32);
  // conv(each, w_each) (mat 0) and conv(notm, w_not) (mat 1)
  for (int k = 0; k < NKIND; ++k) {
    gemm_conv<<<ggrid, gblk, 0, stream>>>(
        EACH_SW + (size_t)k * NPAIR, WSW + 0 * MAT_U32, CE_F32 + (size_t)k * NELEM);
    gemm_conv<<<ggrid, gblk, 0, stream>>>(
        NOTM_SW + (size_t)k * NPAIR, WSW + 1 * MAT_U32, NC_F32 + (size_t)k * NELEM);
  }

  first_combine<<<dim3(NPAIR / 256), dim3(256), 0, stream>>>(
      dots, E_F32, CE_F32, NC_F32, L_F32, S_F32, T_SW);

  for (int d = 0; d < NDEPTH; ++d) {
    for (int k = 0; k < NKIND; ++k) {
      gemm_conv<<<ggrid, gblk, 0, stream>>>(
          T_SW + (size_t)k * NPAIR, WSW + 2 * MAT_U32, CE_F32 + (size_t)k * NELEM);
    }
    depth_combine<<<dim3(NPAIR / 256), dim3(256), 0, stream>>>(
        dots, E_F32, CE_F32, NC_F32, L_F32, S_F32, T_SW);
  }

  mlp1<<<dim3((BATCH * NSQ + 255) / 256), dim3(256), 0, stream>>>(S_F32, W1, X1);
  mlp2<<<dim3((BATCH * NSQ + 255) / 256), dim3(256), 0, stream>>>(X1, W2, X2);
  mlp3<<<dim3((BATCH + 255) / 256), dim3(256), 0, stream>>>(X2, W3, out);
}
